// SpeakerEmbedder_51969104282083
// MI455X (gfx1250) — matrix operand order repaired and checked by reading, not compiled, not run on hardware
//
#include <hip/hip_runtime.h>
#include <hip/hip_bf16.h>

typedef __attribute__((ext_vector_type(16))) _Float16 v16h;
typedef __attribute__((ext_vector_type(8)))  _Float16 v8h;
typedef __attribute__((ext_vector_type(8)))  float    v8f;

#define BATCH   64
#define TOKENS  16384
#define VOCAB   1024
#define HIDDEN  512
#define EMBED   256

// ---------------------------------------------------------------------------
// Weight pack: f32 [K,N] row-major -> f16 in WMMA B-fragment order.
// For each 32x16 (KxN) tile, store 16 contiguous halves per lane:
//   lane = ((krem/8)%2)*16 + n%16 ; h = krem%8 + 8*(krem/16) ; value = W[k][n]
//   dst[(((ntile*(K/32) + kblk)*32) + lane)*16 + h]
// In the GEMM each lane then loads its whole B fragment as one aligned
// 32-byte vector (two global_load_b128) instead of 16 strided b16 loads.
// ---------------------------------------------------------------------------
__global__ void pack_weight_wmma(const float* __restrict__ src,
                                 _Float16* __restrict__ dst, int K, int N) {
    int i = blockIdx.x * blockDim.x + threadIdx.x;
    if (i >= K * N) return;
    int k = i / N, n = i - k * N;
    int ntile = n >> 4, l16 = n & 15;
    int kblk  = k >> 5, krem = k & 31;
    int lane  = (((krem >> 3) & 1) << 4) + l16;    // ((krem/8)%2)*16 + n%16
    int h     = (krem & 7) + ((krem >> 4) << 3);
    size_t o  = ((((size_t)ntile * (K >> 5) + kblk) << 5) + lane) * 16 + h;
    dst[o] = (_Float16)src[i];
}

// ---------------------------------------------------------------------------
// Per-row token histogram via LDS atomics -> pooled f16 activations [64,1024]
// One workgroup (256 threads) per batch row; 1024 bins in LDS (ds_add_u32).
// ---------------------------------------------------------------------------
__global__ void histogram_pool(const int* __restrict__ tokens,
                               _Float16* __restrict__ pooled) {
    __shared__ unsigned cnt[VOCAB];
    const int row = blockIdx.x;
    const int tid = threadIdx.x;

    #pragma unroll
    for (int i = 0; i < VOCAB / 256; ++i) cnt[tid + 256 * i] = 0u;
    __syncthreads();

    const int* tr = tokens + (size_t)row * TOKENS;
    for (int t = tid; t < TOKENS; t += 256) {
        int idx = tr[t] & (VOCAB - 1);   // tokens % 1024 (tokens non-negative)
        atomicAdd(&cnt[idx], 1u);
    }
    __syncthreads();

    const float inv = 1.0f / (float)TOKENS;
    #pragma unroll
    for (int i = 0; i < VOCAB / 256; ++i) {
        int c = tid + 256 * i;
        pooled[(size_t)row * VOCAB + c] = (_Float16)((float)cnt[c] * inv);
    }
}

// ---------------------------------------------------------------------------
// f16 WMMA GEMM: out[M=64, N] = act( A[64,K] @ W[K,N] + bias ).
// One wave per 16x16 output tile; K stepped by 32 per v_wmma_f32_16x16x32_f16.
// A (16x32 f16, ISA 7.12.2): M = lane%16 ; halves 0..7 -> K = base+0..7,
//   halves 8..15 -> K = 16+base+0..7, base = 8*(lane/16)  => two b128 loads.
// B: pre-packed (see pack_weight_wmma)                    => one 32B load.
// C/D (16x16 f32): N = lane%16 ; vgpr v -> M = v + 8*(lane/16).
// ---------------------------------------------------------------------------
template <bool RELU, bool OUT_F16>
__global__ void gemm_wmma_f16(const _Float16* __restrict__ A,
                              const _Float16* __restrict__ Bp,
                              const float* __restrict__ bias,
                              void* __restrict__ out,
                              int N, int K) {
    const int lane  = threadIdx.x & 31;
    const int wave  = (blockIdx.x * (blockDim.x >> 5)) + (threadIdx.x >> 5);
    const int mtile = wave & 3;          // M = 64 -> 4 tiles
    const int ntile = wave >> 2;

    const int l16 = lane & 15;
    const int hi  = lane >> 4;           // 0 or 1
    const int nkb = K >> 5;              // K blocks of 32

    const _Float16* arow  = A + (size_t)(mtile * 16 + l16) * K + hi * 8;
    const _Float16* bbase = Bp + (((size_t)ntile * nkb) * 32 + lane) * 16;

    v8f acc = {};
    #pragma unroll 2
    for (int kblk = 0; kblk < nkb; ++kblk) {
        __builtin_prefetch(arow + (kblk + 1) * 32, 0, 3);
        __builtin_prefetch(bbase + (size_t)(kblk + 1) * 512, 0, 3);

        v8h alo = *(const v8h*)(arow + kblk * 32);        // K = base+0..7
        v8h ahi = *(const v8h*)(arow + kblk * 32 + 16);   // K = 16+base+0..7
        v16h a = __builtin_shufflevector(alo, ahi,
                 0, 1, 2, 3, 4, 5, 6, 7, 8, 9, 10, 11, 12, 13, 14, 15);

        v16h b = *(const v16h*)(bbase + (size_t)kblk * 512); // 32B contiguous

        acc = __builtin_amdgcn_wmma_f32_16x16x32_f16(
            /*neg_a=*/false, a, /*neg_b=*/false, b,
            /*c_mod=*/(short)0, acc, /*reuse_a=*/false, /*reuse_b=*/false);
    }

    const int   n     = ntile * 16 + l16;
    const float bv    = bias[n];
    const int   mbase = mtile * 16 + hi * 8;
    #pragma unroll
    for (int v = 0; v < 8; ++v) {
        float x = acc[v] + bv;
        if (RELU) x = fmaxf(x, 0.0f);
        size_t o = (size_t)(mbase + v) * N + n;
        if (OUT_F16) ((_Float16*)out)[o] = (_Float16)x;
        else         ((float*)out)[o]    = x;
    }
}

// ---------------------------------------------------------------------------
// Row-wise L2 normalize: out = emb / max(||emb||, 1e-12). One block per row.
// ---------------------------------------------------------------------------
__global__ void l2_normalize(const float* __restrict__ emb,
                             float* __restrict__ out) {
    __shared__ float red[EMBED];
    const int row = blockIdx.x;
    const int tid = threadIdx.x;         // blockDim.x == EMBED == 256

    float v = emb[(size_t)row * EMBED + tid];
    red[tid] = v * v;
    __syncthreads();
    for (int s = EMBED / 2; s > 0; s >>= 1) {
        if (tid < s) red[tid] += red[tid + s];
        __syncthreads();
    }
    float norm  = sqrtf(red[0]);
    float scale = 1.0f / fmaxf(norm, 1e-12f);
    out[(size_t)row * EMBED + tid] = v * scale;
}

// ---------------------------------------------------------------------------
// Launcher
// ---------------------------------------------------------------------------
extern "C" void kernel_launch(void* const* d_in, const int* in_sizes, int n_in,
                              void* d_out, int out_size, void* d_ws, size_t ws_size,
                              hipStream_t stream) {
    const int*   tokens = (const int*)  d_in[0];
    const float* W1     = (const float*)d_in[1];
    const float* b1     = (const float*)d_in[2];
    const float* W2     = (const float*)d_in[3];
    const float* b2     = (const float*)d_in[4];
    const float* W3     = (const float*)d_in[5];
    const float* b3     = (const float*)d_in[6];
    float*       outp   = (float*)d_out;

    // Workspace layout (bytes), 256-aligned sections, ~2.1 MB total.
    char* w = (char*)d_ws;
    _Float16* pooledH = (_Float16*)(w + 0);                 // 64*1024*2   = 131072
    _Float16* W1p     = (_Float16*)(w + 131072);            // 1024*512*2  = 1048576
    _Float16* W2p     = (_Float16*)(w + 1179648);           // 512*512*2   = 524288
    _Float16* W3p     = (_Float16*)(w + 1703936);           // 512*256*2   = 262144
    _Float16* h1      = (_Float16*)(w + 1966080);           // 64*512*2    = 65536
    _Float16* h2      = (_Float16*)(w + 2031616);           // 64*512*2    = 65536
    float*    emb     = (float*)   (w + 2097152);           // 64*256*4    = 65536

    // Weight conversion + WMMA-fragment packing (f32 -> f16)
    pack_weight_wmma<<<(VOCAB * HIDDEN + 255) / 256, 256, 0, stream>>>(W1, W1p, VOCAB, HIDDEN);
    pack_weight_wmma<<<(HIDDEN * HIDDEN + 255) / 256, 256, 0, stream>>>(W2, W2p, HIDDEN, HIDDEN);
    pack_weight_wmma<<<(HIDDEN * EMBED  + 255) / 256, 256, 0, stream>>>(W3, W3p, HIDDEN, EMBED);

    // Token histogram -> pooled activations
    histogram_pool<<<BATCH, 256, 0, stream>>>(tokens, pooledH);

    // Layer 1: [64,1024] @ [1024,512] + b1, relu  -> 128 tiles -> 16 blocks x 8 waves
    gemm_wmma_f16<true,  true ><<<16, 256, 0, stream>>>(pooledH, W1p, b1, h1, HIDDEN, VOCAB);
    // Layer 2: [64,512] @ [512,512] + b2, relu
    gemm_wmma_f16<true,  true ><<<16, 256, 0, stream>>>(h1, W2p, b2, h2, HIDDEN, HIDDEN);
    // Layer 3: [64,512] @ [512,256] + b3 (f32 out) -> 64 tiles -> 8 blocks
    gemm_wmma_f16<false, false><<< 8, 256, 0, stream>>>(h2, W3p, b3, emb, EMBED, HIDDEN);

    // L2 normalize rows
    l2_normalize<<<BATCH, EMBED, 0, stream>>>(emb, outp);
}
